// Wave_generator_26706106647250
// MI455X (gfx1250) — compile-verified, matrix operand order checked by reading, not run on hardware
//
#include <hip/hip_runtime.h>
#include <hip/hip_bf16.h>
#include <math.h>

// ---------------- constants ----------------
#define BB    16
#define NN    1024
#define DD    128
#define HH    8
#define MM    (BB*NN)        // 16384 rows
#define KPAD  160            // 129 padded to multiple of 32
#define QKVD  384

typedef _Float16 v16h __attribute__((ext_vector_type(16)));
typedef float    v8f  __attribute__((ext_vector_type(8)));

#define WMMA16(a,b,c) __builtin_amdgcn_wmma_f32_16x16x32_f16(false,(a),false,(b),(short)0,(c),false,false)

// ---------------- optional CDNA5 async global->LDS path ----------------
#if __has_builtin(__builtin_amdgcn_global_load_async_to_lds_b128)
#define HAVE_ASYNC_LDS 1
#else
#define HAVE_ASYNC_LDS 0
#endif

#if HAVE_ASYNC_LDS
#define AS1 __attribute__((address_space(1)))
#define AS3 __attribute__((address_space(3)))
typedef int v4i_ __attribute__((vector_size(16)));
__device__ __forceinline__ void async_g2l_b128(const void* g, void* l) {
    __builtin_amdgcn_global_load_async_to_lds_b128((AS1 v4i_*)g, (AS3 v4i_*)l, 0, 0);
}
__device__ __forceinline__ void wait_async0() {
    asm volatile("s_wait_asynccnt 0" ::: "memory");
}
#endif

union V16U { v16h v; _Float16 h[16]; uint4 q[2]; };

__device__ __forceinline__ v16h load_pair16(const _Float16* p) {
    V16U u;
    u.q[0] = *reinterpret_cast<const uint4*>(p);
    u.q[1] = *reinterpret_cast<const uint4*>(p + 16);
    return u.v;
}
__device__ __forceinline__ v16h load_half8(const _Float16* p) {
    V16U u;
    u.q[0] = *reinterpret_cast<const uint4*>(p);
    u.q[1] = make_uint4(0u, 0u, 0u, 0u);
    return u.v;
}

// ---------------- weight convert (f32 -> f16, optional column pad) ----------------
__global__ void cvtpad_kernel(const float* __restrict__ src, _Float16* __restrict__ dst,
                              int rows, int sc, int dc) {
    int idx = blockIdx.x * blockDim.x + threadIdx.x;
    if (idx >= rows * dc) return;
    int r = idx / dc, c = idx - r * dc;
    dst[idx] = (_Float16)((c < sc) ? src[r * sc + c] : 0.f);
}

// ---------------- sine encoding -> eeig f16 [M, 160] ----------------
__global__ void sine_kernel(const float* __restrict__ ev, _Float16* __restrict__ out) {
    int idx = blockIdx.x * blockDim.x + threadIdx.x;
    if (idx >= MM * KPAD) return;
    int row = idx / KPAD, col = idx - row * KPAD;
    float e = ev[row];
    float val;
    const float c = 0.14391156806f; // ln(10000)/64
    if (col == 0)       val = e;
    else if (col <= 64) { int i = col - 1;  val = __sinf(e * 100.f * __expf(-(float)i * c)); }
    else if (col <= 128){ int i = col - 65; val = __cosf(e * 100.f * __expf(-(float)i * c)); }
    else                val = 0.f;
    out[idx] = (_Float16)val;
}

// ---------------- generic WMMA GEMM: C[M,Nc] = A[M,K] * W^T + bias ----------------
// A f16 row-major [M,K]; W f16 row-major [Nc,K] (so B[k][n] = W[n][k], contiguous per lane)
// MODE 0: Cf32 = v ; 1: Cf32 += v ; 2: Cf16 = v ; 3: Cf16 = gelu(v)
template <int MODE>
__global__ void wmma_gemm_kernel(const _Float16* __restrict__ A,
                                 const _Float16* __restrict__ W,
                                 const float* __restrict__ bias,
                                 float* __restrict__ Cf32,
                                 _Float16* __restrict__ Cf16,
                                 int K, int Nc) {
    const int lane = threadIdx.x & 31;
    const int wid  = threadIdx.x >> 5;
    const int half = lane >> 4;
    const int l15  = lane & 15;
    const int m0   = blockIdx.x * 16;
    const int n0   = (blockIdx.y * 4 + wid) * 16;

    const _Float16* pA = A + (size_t)(m0 + l15) * K + half * 8;
    const _Float16* pB = W + (size_t)(n0 + l15) * K + half * 8;

    v8f acc = {};
    for (int kc = 0; kc < K; kc += 32) {
        if (kc + 32 < K) __builtin_prefetch((const void*)(pA + kc + 32), 0, 1);
        v16h a = load_pair16(pA + kc);
        v16h b = load_pair16(pB + kc);
        acc = WMMA16(a, b, acc);
    }
    const float bs  = bias[n0 + l15];
    const int   col = n0 + l15;
    #pragma unroll
    for (int j = 0; j < 8; ++j) {
        int row = m0 + j + half * 8;
        float v = acc[j] + bs;
        size_t o = (size_t)row * Nc + col;
        if (MODE == 0)      Cf32[o] = v;
        else if (MODE == 1) Cf32[o] += v;
        else if (MODE == 2) Cf16[o] = (_Float16)v;
        else {
            float g = 0.5f * v * (1.f + erff(v * 0.70710678118f)); // exact gelu
            Cf16[o] = (_Float16)g;
        }
    }
}

// ---------------- LayerNorm (f32 in, f16 out), one row per 128-thread block ----------------
__global__ void ln_kernel(const float* __restrict__ x, const float* __restrict__ sc,
                          const float* __restrict__ bi, _Float16* __restrict__ out) {
    const int row = blockIdx.x, t = threadIdx.x;
    const int wid = t >> 5, lane = t & 31;
    float v = x[(size_t)row * DD + t];
    float sum = v;
    for (int off = 16; off; off >>= 1) sum += __shfl_xor(sum, off, 32);
    __shared__ float sm[4], sv[4];
    if (lane == 0) sm[wid] = sum;
    __syncthreads();
    float mean = (sm[0] + sm[1] + sm[2] + sm[3]) * (1.f / 128.f);
    float d = v - mean;
    float vs = d * d;
    for (int off = 16; off; off >>= 1) vs += __shfl_xor(vs, off, 32);
    if (lane == 0) sv[wid] = vs;
    __syncthreads();
    float var = (sv[0] + sv[1] + sv[2] + sv[3]) * (1.f / 128.f);
    float r = rsqrtf(var + 1e-5f);
    out[(size_t)row * DD + t] = (_Float16)(d * r * sc[t] + bi[t]);
}

// ---------------- flash attention: 1 wave = 16 queries of one (b,h) ----------------
// qkv f16 [M, 384]: q @ col 0, k @ 128, v @ 256; head h at +h*16, dh = 16 (K padded to 32)
__global__ void attn_kernel(const _Float16* __restrict__ qkv, const int* __restrict__ length,
                            _Float16* __restrict__ mo) {
    __shared__ _Float16 Pl[4][16][32];
    __shared__ _Float16 Vl[4][32][16];

    const int lane = threadIdx.x & 31;
    const int wid  = threadIdx.x >> 5;
    const int half = lane >> 4;
    const int l15  = lane & 15;
    const int b = blockIdx.z, h = blockIdx.y;
    const int qt = blockIdx.x * 4 + wid;            // 0..63
    const int len = length[b];

    const size_t rowbase = (size_t)b * NN;
    const _Float16* qp = qkv + (rowbase + qt * 16 + l15) * QKVD + h * 16 + half * 8;
    v16h qa = load_half8(qp);

    float m_[8], l_[8];
    v8f o = {};
    #pragma unroll
    for (int j = 0; j < 8; ++j) { m_[j] = -1e30f; l_[j] = 0.f; }

    for (int kt = 0; kt < NN; kt += 32) {
        // --- scores: S = Q K^T for 32 keys (two 16-col WMMAs) ---
        const _Float16* kp0 = qkv + (rowbase + kt + l15) * QKVD + 128 + h * 16 + half * 8;
        v16h kf0 = load_half8(kp0);
        v16h kf1 = load_half8(kp0 + 16 * QKVD);
        v8f s0 = {}, s1 = {};
        s0 = WMMA16(qa, kf0, s0);
        s1 = WMMA16(qa, kf1, s1);

        // --- stage V tile (32 keys x 16 dh) into wave-private LDS ---
        // Async global->LDS copy overlaps with the softmax VALU work below.
        const _Float16* vp = qkv + (rowbase + kt + lane) * QKVD + 256 + h * 16;
#if HAVE_ASYNC_LDS
        async_g2l_b128(vp,     &Vl[wid][lane][0]);
        async_g2l_b128(vp + 8, &Vl[wid][lane][8]);
#else
        {
            uint4 v0 = *reinterpret_cast<const uint4*>(vp);
            uint4 v1 = *reinterpret_cast<const uint4*>(vp + 8);
            *reinterpret_cast<uint4*>(&Vl[wid][lane][0]) = v0;
            *reinterpret_cast<uint4*>(&Vl[wid][lane][8]) = v1;
        }
#endif

        const bool mask0 = (kt + l15) >= len;
        const bool mask1 = (kt + 16 + l15) >= len;

        float p0[8], p1[8];
        #pragma unroll
        for (int j = 0; j < 8; ++j) {
            float a0 = mask0 ? -1e30f : s0[j] * 0.25f;   // 1/sqrt(16)
            float a1 = mask1 ? -1e30f : s1[j] * 0.25f;
            float mx = fmaxf(a0, a1);
            mx = fmaxf(mx, __shfl_xor(mx, 1, 32));
            mx = fmaxf(mx, __shfl_xor(mx, 2, 32));
            mx = fmaxf(mx, __shfl_xor(mx, 4, 32));
            mx = fmaxf(mx, __shfl_xor(mx, 8, 32));
            float mn = fmaxf(m_[j], mx);
            float e0 = __expf(a0 - mn);
            float e1 = __expf(a1 - mn);
            float rs = e0 + e1;
            rs += __shfl_xor(rs, 1, 32);
            rs += __shfl_xor(rs, 2, 32);
            rs += __shfl_xor(rs, 4, 32);
            rs += __shfl_xor(rs, 8, 32);
            float al = __expf(m_[j] - mn);
            l_[j] = l_[j] * al + rs;
            m_[j] = mn;
            o[j] *= al;
            p0[j] = e0; p1[j] = e1;
            // stage P (row = j+8*half, cols l15 / 16+l15) into wave-private LDS
            Pl[wid][j + half * 8][l15]      = (_Float16)e0;
            Pl[wid][j + half * 8][16 + l15] = (_Float16)e1;
        }

        // --- A-frag of P: row = l15, K chunks per ISA 16-bit layout (k = 8*half+e, then 16+8*half+e) ---
        V16U pu;
        pu.q[0] = *reinterpret_cast<const uint4*>(&Pl[wid][l15][half * 8]);
        pu.q[1] = *reinterpret_cast<const uint4*>(&Pl[wid][l15][16 + half * 8]);

#if HAVE_ASYNC_LDS
        wait_async0();   // V tile is now resident in LDS
#endif
        // --- B-frag of V: col = l15 (dh), keys per ISA layout ---
        V16U vu;
        #pragma unroll
        for (int e = 0; e < 8; ++e)  vu.h[e]     = Vl[wid][half * 8 + e][l15];
        #pragma unroll
        for (int e = 0; e < 8; ++e)  vu.h[8 + e] = Vl[wid][16 + half * 8 + e][l15];

        o = WMMA16(pu.v, vu.v, o);   // O = alpha*O + P*V
    }

    // epilogue: O /= l, write mo[b,n,h*16+dh]
    #pragma unroll
    for (int j = 0; j < 8; ++j) {
        float val = o[j] / l_[j];
        int row = qt * 16 + j + half * 8;
        mo[(rowbase + row) * DD + h * 16 + l15] = (_Float16)val;
    }
}

// ---------------- masked pooled decoder heads -> coe[b][20] ----------------
// rows 0..7 = sca, 8..15 = wav, 16..19 = scl
__global__ void pooled_kernel(const float* __restrict__ eig,
                              const float* __restrict__ wsca, const float* __restrict__ bsca,
                              const float* __restrict__ wwav, const float* __restrict__ bwav,
                              const float* __restrict__ wscl, const float* __restrict__ bscl,
                              const int* __restrict__ length, float* __restrict__ coe) {
    __shared__ float sw[20 * 128];
    __shared__ float sacc[20];
    const int b = blockIdx.x, t = threadIdx.x;
#if HAVE_ASYNC_LDS
    // async-stage the 20x128 f32 weight block into LDS in 16B chunks
    for (int c = t; c < 640; c += 256) {
        int f0 = c * 4;
        int j = f0 >> 7, k = f0 & 127;
        const float* srcp = (j < 8) ? (wsca + j * 128 + k)
                          : (j < 16) ? (wwav + (j - 8) * 128 + k)
                                     : (wscl + (j - 16) * 128 + k);
        async_g2l_b128(srcp, &sw[f0]);
    }
    wait_async0();
#else
    for (int i = t; i < 20 * 128; i += 256) {
        int j = i >> 7, k = i & 127;
        sw[i] = (j < 8) ? wsca[j * 128 + k] : (j < 16) ? wwav[(j - 8) * 128 + k]
                                                       : wscl[(j - 16) * 128 + k];
    }
#endif
    if (t < 20) sacc[t] = 0.f;
    __syncthreads();

    const int len = length[b];
    float p[20];
    #pragma unroll
    for (int j = 0; j < 20; ++j) p[j] = 0.f;
    for (int n = t; n < len; n += 256) {
        const float* row = eig + ((size_t)b * NN + n) * DD;
        for (int k = 0; k < 128; ++k) {
            float xv = row[k];
            #pragma unroll
            for (int j = 0; j < 20; ++j) p[j] += xv * sw[j * 128 + k];
        }
    }
    #pragma unroll
    for (int j = 0; j < 20; ++j) atomicAdd(&sacc[j], p[j]);
    __syncthreads();

    if (t == 0) {
        float lenf = (float)len;
        float sg[20];
        for (int j = 0; j < 20; ++j) {
            float bj = (j < 8) ? bsca[j] : (j < 16) ? bwav[j - 8] : bscl[j - 16];
            float v = (sacc[j] + lenf * bj) / (lenf + 1e-8f);
            sg[j] = 1.f / (1.f + __expf(-v));
        }
        float ss = 0.f, sv = 0.f;
        for (int j = 0; j < 8; ++j)  ss += sg[j];
        for (int j = 8; j < 16; ++j) sv += sg[j];
        for (int j = 0; j < 8; ++j)  coe[b * 20 + j]      = sg[j] / (ss + 1e-8f);
        for (int j = 0; j < 8; ++j)  coe[b * 20 + 8 + j]  = sg[8 + j] / (sv + 1e-8f);
        for (int j = 0; j < 4; ++j)  coe[b * 20 + 16 + j] = sg[16 + j] * 2.0f; // PRE_S
    }
}

// ---------------- Chebyshev bases + tight-frame normalize -> out[B,N,5] ----------------
__global__ void final_kernel(const float* __restrict__ ev, const float* __restrict__ coe,
                             float* __restrict__ out) {
    int idx = blockIdx.x * blockDim.x + threadIdx.x;
    if (idx >= MM) return;
    int b = idx >> 10;
    float e = ev[idx];
    const float* c = coe + b * 20;

    // scaling: sum_k coe_sca[k] * 0.5*(1 - T_{2k+1}(e-1))
    float y = e - 1.f, te = 1.f, to = y;
    float s = c[0] * 0.5f * (1.f - to);
    #pragma unroll
    for (int k = 1; k < 8; ++k) {
        te = 2.f * y * to - te;
        to = 2.f * y * te - to;
        s += c[k] * 0.5f * (1.f - to);
    }
    // wavelet: per j, sum_k coe_wav[k] * 0.5*(1 - T_{2k}(fsw-1))
    float w[4];
    #pragma unroll
    for (int j = 0; j < 4; ++j) {
        float f = e * c[16 + j];
        if (f > 2.f) f = 0.f;
        float yw = f - 1.f, te2 = 1.f, to2 = yw;
        float acc = c[8] * 0.5f * (1.f - te2);
        #pragma unroll
        for (int k = 1; k < 8; ++k) {
            te2 = 2.f * yw * to2 - te2;
            to2 = 2.f * yw * te2 - to2;
            acc += c[8 + k] * 0.5f * (1.f - te2);
        }
        w[j] = acc;
    }
    float nrm = s * s;
    #pragma unroll
    for (int j = 0; j < 4; ++j) nrm += w[j] * w[j];
    float inv = 1.f / (sqrtf(nrm) + 1e-8f);
    float* op = out + (size_t)idx * 5;
    op[0] = s * inv;
    #pragma unroll
    for (int j = 0; j < 4; ++j) op[1 + j] = w[j] * inv;
}

// ---------------- host orchestration ----------------
static inline int cdiv(int a, int b) { return (a + b - 1) / b; }

extern "C" void kernel_launch(void* const* d_in, const int* in_sizes, int n_in,
                              void* d_out, int out_size, void* d_ws, size_t ws_size,
                              hipStream_t stream) {
    (void)in_sizes; (void)n_in; (void)out_size; (void)ws_size;
    const float* eigenvalue = (const float*)d_in[1];
    const int*   length     = (const int*)d_in[2];
    const float* eig_w_W = (const float*)d_in[3];
    const float* eig_w_b = (const float*)d_in[4];
    const float* ln1_s = (const float*)d_in[5];
    const float* ln1_b = (const float*)d_in[6];
    const float* ln2_s = (const float*)d_in[7];
    const float* ln2_b = (const float*)d_in[8];
    const float* qkv_W = (const float*)d_in[9];
    const float* qkv_b = (const float*)d_in[10];
    const float* out_W = (const float*)d_in[11];
    const float* out_b = (const float*)d_in[12];
    const float* ffn_W1 = (const float*)d_in[13];
    const float* ffn_b1 = (const float*)d_in[14];
    const float* ffn_W2 = (const float*)d_in[15];
    const float* ffn_b2 = (const float*)d_in[16];
    const float* dec_sca_W = (const float*)d_in[17];
    const float* dec_sca_b = (const float*)d_in[18];
    const float* dec_wav_W = (const float*)d_in[19];
    const float* dec_wav_b = (const float*)d_in[20];
    const float* dec_scl_W = (const float*)d_in[21];
    const float* dec_scl_b = (const float*)d_in[22];
    float* outp = (float*)d_out;

    char* ws = (char*)d_ws;
    float*     eig_f32 = (float*)(ws);                         //  8,388,608 B
    _Float16*  f16A    = (_Float16*)(ws + 8388608);            //  5,242,880 B (eeig, later gelu-out)
    _Float16*  f16B    = (_Float16*)(ws + 13631488);           //  4,194,304 B (ln outputs)
    _Float16*  qkv16   = (_Float16*)(ws + 17825792);           // 12,582,912 B
    _Float16*  mo16    = (_Float16*)(ws + 30408704);           //  4,194,304 B
    _Float16*  eigW16  = (_Float16*)(ws + 34603008);           //     40,960 B
    _Float16*  qkvW16  = (_Float16*)(ws + 34643968);           //     98,304 B
    _Float16*  outW16  = (_Float16*)(ws + 34742272);           //     32,768 B
    _Float16*  ffn1W16 = (_Float16*)(ws + 34775040);           //     32,768 B
    _Float16*  ffn2W16 = (_Float16*)(ws + 34807808);           //     32,768 B
    float*     coe     = (float*)(ws + 34840576);              //      1,280 B

    // 1) weight conversions
    cvtpad_kernel<<<cdiv(128 * KPAD, 256), 256, 0, stream>>>(eig_w_W, eigW16, 128, 129, KPAD);
    cvtpad_kernel<<<cdiv(QKVD * DD, 256), 256, 0, stream>>>(qkv_W, qkvW16, QKVD, DD, DD);
    cvtpad_kernel<<<cdiv(DD * DD, 256), 256, 0, stream>>>(out_W, outW16, DD, DD, DD);
    cvtpad_kernel<<<cdiv(DD * DD, 256), 256, 0, stream>>>(ffn_W1, ffn1W16, DD, DD, DD);
    cvtpad_kernel<<<cdiv(DD * DD, 256), 256, 0, stream>>>(ffn_W2, ffn2W16, DD, DD, DD);

    // 2) sine encoding -> f16A [M,160]
    sine_kernel<<<cdiv(MM * KPAD, 256), 256, 0, stream>>>(eigenvalue, f16A);

    // 3) eig = eeig @ eigW^T + b   -> eig_f32
    wmma_gemm_kernel<0><<<dim3(MM / 16, DD / 64), 128, 0, stream>>>(f16A, eigW16, eig_w_b,
                                                                    eig_f32, nullptr, KPAD, DD);
    // 4) ln1 -> f16B
    ln_kernel<<<MM, 128, 0, stream>>>(eig_f32, ln1_s, ln1_b, f16B);
    // 5) qkv = h @ qkvW^T + b  -> qkv16 [M,384]
    wmma_gemm_kernel<2><<<dim3(MM / 16, QKVD / 64), 128, 0, stream>>>(f16B, qkvW16, qkv_b,
                                                                      nullptr, qkv16, DD, QKVD);
    // 6) flash attention -> mo16
    attn_kernel<<<dim3(NN / 64, HH, BB), 128, 0, stream>>>(qkv16, length, mo16);
    // 7) eig += mo @ outW^T + b
    wmma_gemm_kernel<1><<<dim3(MM / 16, DD / 64), 128, 0, stream>>>(mo16, outW16, out_b,
                                                                    eig_f32, nullptr, DD, DD);
    // 8) ln2 -> f16B
    ln_kernel<<<MM, 128, 0, stream>>>(eig_f32, ln2_s, ln2_b, f16B);
    // 9) t = gelu(f @ W1^T + b1) -> f16A
    wmma_gemm_kernel<3><<<dim3(MM / 16, DD / 64), 128, 0, stream>>>(f16B, ffn1W16, ffn_b1,
                                                                    nullptr, f16A, DD, DD);
    // 10) eig += t @ W2^T + b2
    wmma_gemm_kernel<1><<<dim3(MM / 16, DD / 64), 128, 0, stream>>>(f16A, ffn2W16, ffn_b2,
                                                                    eig_f32, nullptr, DD, DD);
    // 11) pooled decoder heads -> coe[b][20]
    pooled_kernel<<<BB, 256, 0, stream>>>(eig_f32, dec_sca_W, dec_sca_b, dec_wav_W, dec_wav_b,
                                          dec_scl_W, dec_scl_b, length, coe);
    // 12) Chebyshev + tight frames -> out
    final_kernel<<<cdiv(MM, 256), 256, 0, stream>>>(eigenvalue, coe, outp);
}
